// MyModel_87454124082185
// MI455X (gfx1250) — compile-verified, matrix operand order checked by reading, not run on hardware
//
#include <hip/hip_runtime.h>
#include <hip/hip_bf16.h>

// ---------------- problem constants ----------------
#define BB   8          // batch
#define TT   16         // timesteps
#define HIN  640
#define WIN  480
#define HP   320        // pooled H
#define WP   240        // pooled W
#define HC   318        // conv-valid H
#define WC   238        // conv-valid W
#define HW   (HC*WC)    // 75684 pixels per image
#define FF   16         // ConvLSTM filters
#define NG   64         // 4*F gate channels
#define KPAD 160        // K = 9*16 = 144 padded to 5 chunks of 32
#define TILES_PER_IMG ((HW + 15) / 16)        // 4731
#define NTILES (BB * TILES_PER_IMG)           // 37848
#define FLAT (HW * FF)                        // 1210944

typedef __attribute__((ext_vector_type(16))) __bf16 v16bf;
typedef __attribute__((ext_vector_type(8)))  float  v8f;

union Frag { v16bf v; uint4 u[2]; };

// ---------------- bf16 helpers ----------------
__device__ __forceinline__ float b2f(unsigned short h) {
    return __uint_as_float(((unsigned)h) << 16);
}
__device__ __forceinline__ unsigned short f2b(float f) {
    unsigned u = __float_as_uint(f);
    unsigned r = u + 0x7FFFu + ((u >> 16) & 1u);   // round-to-nearest-even
    return (unsigned short)(r >> 16);
}
__device__ __forceinline__ float hsig(float x) {
    return fminf(fmaxf(x * 0.2f + 0.5f, 0.0f), 1.0f);
}

// ---------------- 0) zero scratch (uint4 granularity) ----------------
__global__ void zero_kernel(uint4* __restrict__ p, int n16) {
    int i = blockIdx.x * blockDim.x + threadIdx.x;
    if (i < n16) { uint4 z; z.x = z.y = z.z = z.w = 0u; p[i] = z; }
}

// ---------------- 1) weight prep: Bt[n][k] bf16, K padded ----------------
__global__ void prep_weights_kernel(const float* __restrict__ rk,
                                    unsigned short* __restrict__ Bt) {
    int i = blockIdx.x * blockDim.x + threadIdx.x;     // over 64*160
    if (i >= NG * KPAD) return;
    int n = i / KPAD, k = i - n * KPAD;
    float v = (k < 144) ? rk[k * NG + n] : 0.0f;       // rec_kernel (3,3,16,64) flat
    Bt[i] = f2b(v);
}

// ---------------- 2) maxpool 2x2 for timestep t ----------------
__global__ void pool_kernel(const float* __restrict__ in, float* __restrict__ pooled, int t) {
    int idx = blockIdx.x * blockDim.x + threadIdx.x;   // B*HP*WP
    if (idx >= BB * HP * WP) return;
    int b = idx / (HP * WP);
    int r = idx - b * (HP * WP);
    int y = r / WP, x = r - y * WP;
    const float* p = in + ((((size_t)b * TT + t) * HIN + 2 * y) * WIN + 2 * x);
    pooled[idx] = fmaxf(fmaxf(p[0], p[1]), fmaxf(p[WIN], p[WIN + 1]));
}

// ---------------- 3) input conv 3x3 valid, C=1 -> 64, +bias, bf16 out ----------------
__global__ void xconv_kernel(const float* __restrict__ pooled,
                             const float* __restrict__ w,      // (3,3,1,64) flat: w[t9*64+n]
                             const float* __restrict__ bias,
                             unsigned short* __restrict__ xc) {
    int idx = blockIdx.x * blockDim.x + threadIdx.x;   // B*HW pixels
    if (idx >= BB * HW) return;
    int b = idx / HW;
    int p = idx - b * HW;
    int y = p / WC, x = p - y * WC;
    const float* pb = pooled + ((size_t)b * HP + y) * WP + x;
    float v[9];
#pragma unroll
    for (int dy = 0; dy < 3; ++dy)
#pragma unroll
        for (int dx = 0; dx < 3; ++dx) v[dy * 3 + dx] = pb[(size_t)dy * WP + dx];

    unsigned short* o = xc + (size_t)idx * NG;
#pragma unroll
    for (int n0 = 0; n0 < NG; n0 += 8) {
        unsigned tmp[4];
#pragma unroll
        for (int j = 0; j < 4; ++j) {
            float a0 = bias[n0 + 2 * j], a1 = bias[n0 + 2 * j + 1];
#pragma unroll
            for (int t9 = 0; t9 < 9; ++t9) {
                a0 = fmaf(v[t9], w[t9 * NG + n0 + 2 * j],     a0);
                a1 = fmaf(v[t9], w[t9 * NG + n0 + 2 * j + 1], a1);
            }
            tmp[j] = (unsigned)f2b(a0) | ((unsigned)f2b(a1) << 16);
        }
        uint4 pk; pk.x = tmp[0]; pk.y = tmp[1]; pk.z = tmp[2]; pk.w = tmp[3];
        *(uint4*)(o + n0) = pk;
    }
}

// ---------------- 4) LSTM step: recurrent conv as implicit-GEMM WMMA ----------------
// A-frag (16x32 bf16): lane l -> pixel m=l&15; 16 K-values = 8 channels of tap 2c
// and 8 channels of tap 2c+1, channel base = (l>>4)*8 -> two 16B contiguous loads.
__device__ __forceinline__ uint4 load_tap(const unsigned short* __restrict__ hb,
                                          int y, int x, int tap, int ch0, bool valid) {
    uint4 r; r.x = r.y = r.z = r.w = 0u;
    int dy = tap / 3 - 1, dx = tap % 3 - 1;
    int yy = y + dy, xx = x + dx;
    if (valid && yy >= 0 && yy < HC && xx >= 0 && xx < WC)
        r = *(const uint4*)(hb + ((size_t)yy * WC + xx) * FF + ch0);
    return r;
}

__global__ void __launch_bounds__(128)
lstm_step_kernel(const unsigned short* __restrict__ hprev,   // bf16 (B,HC,WC,F)
                 unsigned short* __restrict__ hnext,         // bf16 (B,HC,WC,F)
                 float* __restrict__ cbuf,                    // f32  (B,HC,WC,F)
                 const unsigned short* __restrict__ xconv,   // bf16 (B,HW,64)
                 const unsigned short* __restrict__ Bt) {    // bf16 (64,160)
    __shared__ __align__(16) unsigned short ldsW[NG * KPAD]; // 20 KB
    {   // cooperative stage of recurrent weights into LDS
        const unsigned* src = (const unsigned*)Bt;
        unsigned* dst = (unsigned*)ldsW;
        for (int i = threadIdx.x; i < NG * KPAD / 2; i += 128) dst[i] = src[i];
    }
    __syncthreads();

    int wid  = threadIdx.x >> 5;
    int lane = threadIdx.x & 31;
    int tile = blockIdx.x * 4 + wid;
    if (tile >= NTILES) return;                 // wave-uniform: EXEC stays all-1s

    int b  = tile / TILES_PER_IMG;
    int tp = tile - b * TILES_PER_IMG;
    int p0 = tp * 16;

    int m     = lane & 15;
    int chsel = lane >> 4;
    int ch0   = chsel << 3;
    int p     = p0 + m;
    bool valid = (p < HW);
    int y = p / WC, x = p - y * WC;
    const unsigned short* hb = hprev + (size_t)b * HW * FF;

    v8f acc[4];
#pragma unroll
    for (int nt = 0; nt < 4; ++nt) acc[nt] = (v8f)0.0f;

#pragma unroll
    for (int c = 0; c < 5; ++c) {               // K chunks of 32 (taps 2c, 2c+1)
        Frag A;
        A.u[0] = load_tap(hb, y, x, 2 * c, ch0, valid);
        if (2 * c + 1 < 9) A.u[1] = load_tap(hb, y, x, 2 * c + 1, ch0, valid);
        else { A.u[1].x = A.u[1].y = A.u[1].z = A.u[1].w = 0u; }

        int k0 = c * 32 + ch0;
#pragma unroll
        for (int nt = 0; nt < 4; ++nt) {        // N tiles = gates i,f,c,o
            Frag Bf;
            int n = nt * 16 + (lane & 15);
            Bf.u[0] = *(const uint4*)(&ldsW[n * KPAD + k0]);
            Bf.u[1] = *(const uint4*)(&ldsW[n * KPAD + k0 + 16]);
            acc[nt] = __builtin_amdgcn_wmma_f32_16x16x32_bf16(
                false, A.v, false, Bf.v, (short)0, acc[nt], false, false);
        }
    }

    // Epilogue: gates + state update. C/D layout: VGPR r -> pixel r + 8*(lane>=16),
    // channel-in-gate = lane&15.
    int nl = lane & 15;
#pragma unroll
    for (int r = 0; r < 8; ++r) {
        int mp = r + (chsel << 3);
        int pp = p0 + mp;
        if (pp < HW) {
            size_t pix = (size_t)b * HW + pp;
            const unsigned short* xcp = xconv + pix * NG + nl;
            float zi = acc[0][r] + b2f(xcp[0]);
            float zf = acc[1][r] + b2f(xcp[16]);
            float zc = acc[2][r] + b2f(xcp[32]);
            float zo = acc[3][r] + b2f(xcp[48]);
            float gi = hsig(zi), gf = hsig(zf), go = hsig(zo);
            size_t ci = pix * FF + nl;
            float cn = gf * cbuf[ci] + gi * tanhf(zc);
            cbuf[ci]  = cn;
            hnext[ci] = f2b(go * tanhf(cn));
        }
    }
}

// ---------------- 5) dense head: out[b] = h[b,:] . w + bias ----------------
__global__ void dense_kernel(const unsigned short* __restrict__ h,
                             const float* __restrict__ w,
                             const float* __restrict__ bias,
                             float* __restrict__ out) {
    int b = blockIdx.x;
    const unsigned short* hb = h + (size_t)b * FLAT;
    float s = 0.0f;
    for (int i = threadIdx.x; i < FLAT; i += 256)
        s = fmaf(b2f(hb[i]), w[i], s);
    __shared__ float red[256];
    red[threadIdx.x] = s;
    __syncthreads();
    for (int o = 128; o > 0; o >>= 1) {
        if (threadIdx.x < o) red[threadIdx.x] += red[threadIdx.x + o];
        __syncthreads();
    }
    if (threadIdx.x == 0) out[b] = red[0] + bias[0];
}

// ---------------- launcher ----------------
extern "C" void kernel_launch(void* const* d_in, const int* in_sizes, int n_in,
                              void* d_out, int out_size, void* d_ws, size_t ws_size,
                              hipStream_t stream) {
    const float* inputs = (const float*)d_in[0];   // (8,16,640,480,1)
    const float* kern   = (const float*)d_in[1];   // (3,3,1,64)
    const float* rk     = (const float*)d_in[2];   // (3,3,16,64)
    const float* bias   = (const float*)d_in[3];   // (64,)
    const float* dw     = (const float*)d_in[4];   // (flat,1)
    const float* db     = (const float*)d_in[5];   // (1,)
    float* out = (float*)d_out;                    // (8,1)

    char* ws = (char*)d_ws;
    size_t off = 0;
    auto take = [&](size_t bytes) { char* p = ws + off; off = (off + bytes + 255) & ~(size_t)255; return p; };

    float*          pooled = (float*)         take((size_t)BB * HP * WP * 4);        // per-step
    unsigned short* xconv  = (unsigned short*)take((size_t)BB * HW * NG * 2);        // per-step bf16
    unsigned short* h0     = (unsigned short*)take((size_t)BB * HW * FF * 2);
    unsigned short* h1     = (unsigned short*)take((size_t)BB * HW * FF * 2);
    float*          cbuf   = (float*)         take((size_t)BB * HW * FF * 4);
    unsigned short* Bt     = (unsigned short*)take((size_t)NG * KPAD * 2);

    // weight prep + zero init (h0, cbuf) every call (deterministic)
    prep_weights_kernel<<<(NG * KPAD + 255) / 256, 256, 0, stream>>>(rk, Bt);
    {
        int n16 = (int)((size_t)BB * HW * FF * 2 / 16);
        zero_kernel<<<(n16 + 255) / 256, 256, 0, stream>>>((uint4*)h0, n16);
        int n16c = (int)((size_t)BB * HW * FF * 4 / 16);
        zero_kernel<<<(n16c + 255) / 256, 256, 0, stream>>>((uint4*)cbuf, n16c);
    }

    unsigned short* hbufs[2] = { h0, h1 };
    for (int t = 0; t < TT; ++t) {
        pool_kernel <<<(BB * HP * WP + 255) / 256, 256, 0, stream>>>(inputs, pooled, t);
        xconv_kernel<<<(BB * HW + 255) / 256,      256, 0, stream>>>(pooled, kern, bias, xconv);
        lstm_step_kernel<<<(NTILES + 3) / 4, 128, 0, stream>>>(
            hbufs[t & 1], hbufs[(t + 1) & 1], cbuf, xconv, Bt);
    }
    // t=15 wrote hbufs[0]
    dense_kernel<<<BB, 256, 0, stream>>>(hbufs[0], dw, db, out);
}
